// OA_2_33097017983622
// MI455X (gfx1250) — matrix operand order repaired and checked by reading, not compiled, not run on hardware
//
#include <hip/hip_runtime.h>
#include <hip/hip_bf16.h>

// Problem constants (match reference)
#define B_   8
#define C_   512
#define N_   2048
#define C2_  256
#define BN_EPS 1e-5f

typedef __attribute__((ext_vector_type(16))) __bf16 v16bf;
typedef __attribute__((ext_vector_type(8)))  __bf16 v8bf;
typedef __attribute__((ext_vector_type(8)))  float  v8f;

// ---------------------------------------------------------------------------
// Fragment loaders: operands are bf16 with K-stride 1 (guaranteed by the
// pipeline's layout choices), so each fragment is two global_load_b128.
// A-frag (16x32): lane l, half j -> K = j + 8*((j>=8)+(l>=16))
//   j=0..7  contiguous at K = k0 + 8*lhi
//   j=8..15 contiguous at K = k0 + 8*lhi + 16
// B-frag (32x16): same K mapping as A-frag.
// ---------------------------------------------------------------------------
static __device__ __forceinline__ v16bf load_afrag(const __bf16* __restrict__ p,
                                                   int k0, int lhi) {
  const __bf16* q = p + k0 + 8 * lhi;
  v8bf lo = *(const v8bf*)q;
  v8bf hi = *(const v8bf*)(q + 16);
  return __builtin_shufflevector(lo, hi, 0,1,2,3,4,5,6,7,8,9,10,11,12,13,14,15);
}
static __device__ __forceinline__ v16bf load_bfrag(const __bf16* __restrict__ p,
                                                   int k0, int lhi) {
  const __bf16* q = p + k0 + 8 * lhi;
  v8bf lo = *(const v8bf*)q;
  v8bf hi = *(const v8bf*)(q + 16);
  return __builtin_shufflevector(lo, hi, 0,1,2,3,4,5,6,7,8,9,10,11,12,13,14,15);
}

// ---------------------------------------------------------------------------
// GEMM: C[b] = epilogue( A[b] (MxK) * B[b] (KxN) ), all operands bf16, KS==1.
// A element (m,k)  at m*ARS + k  + b*ABS
// B element (k,nc) at nc*BCS + k + b*BBS     (i.e. stored N-major, K contig)
// One wave: 32x64 tile = 2 A-frags x 4 B-frags = 8 WMMA per 32-wide K step.
// Block 128 threads = 4 waves -> 32x256 tile.
// MODE: 0 = bf16 store, 1 = +bias bf16 store, 2 = f32 store,
//       3 = d = x - acc*cinv[col] (bf16), 4 = bias+BN+ReLU+residual (f32).
// CT:   store transposed: element (row,col) at col*CRS + row (+b*CBS);
//       per-lane 8 consecutive rows -> single 16B v8bf store (modes 0/3).
// ---------------------------------------------------------------------------
template<int MODE, bool CT,
         long long ARS, long long ABS,
         long long BCS, long long BBS,
         long long CRS, long long CBS,
         long long XRS, long long XBS,
         int KDIM>
__global__ __launch_bounds__(128) void oa_gemm(
    const __bf16* __restrict__ Am, const __bf16* __restrict__ Bm, void* __restrict__ Cm,
    const float* __restrict__ bias, const float* __restrict__ cinv,
    const float* __restrict__ xres,
    const float* __restrict__ gamma, const float* __restrict__ beta,
    const float* __restrict__ mean, const float* __restrict__ var)
{
  const int lane = threadIdx.x & 31;
  const int wave = threadIdx.x >> 5;
  const int lm   = lane & 15;
  const int lhi  = lane >> 4;
  const int m0 = blockIdx.x * 32;
  const int n0 = blockIdx.y * 256 + wave * 64;
  const int b  = blockIdx.z;

  v8f acc[2][4];
#pragma unroll
  for (int mt = 0; mt < 2; ++mt)
#pragma unroll
    for (int t = 0; t < 4; ++t)
#pragma unroll
      for (int v = 0; v < 8; ++v) acc[mt][t][v] = 0.0f;

  const __bf16* a0p = Am + (long long)(m0 + lm) * ARS + (long long)b * ABS;
  const __bf16* a1p = a0p + 16 * ARS;
  const __bf16* bp[4];
#pragma unroll
  for (int t = 0; t < 4; ++t)
    bp[t] = Bm + (long long)(n0 + t * 16 + lm) * BCS + (long long)b * BBS;

  for (int k0 = 0; k0 < KDIM; k0 += 32) {
    const v16bf a0 = load_afrag(a0p, k0, lhi);
    const v16bf a1 = load_afrag(a1p, k0, lhi);
#pragma unroll
    for (int t = 0; t < 4; ++t) {
      const v16bf bf = load_bfrag(bp[t], k0, lhi);
      acc[0][t] = __builtin_amdgcn_wmma_f32_16x16x32_bf16(
          false, a0, false, bf, (short)0, acc[0][t], false, false);
      acc[1][t] = __builtin_amdgcn_wmma_f32_16x16x32_bf16(
          false, a1, false, bf, (short)0, acc[1][t], false, false);
    }
  }

  // C/D layout: lane l, vgpr v -> row = v + 8*(l>=16), col = l&15
#pragma unroll
  for (int mt = 0; mt < 2; ++mt) {
#pragma unroll
    for (int t = 0; t < 4; ++t) {
      const int col = n0 + t * 16 + lm;
      const int rbase = m0 + mt * 16 + 8 * lhi;
      if constexpr (CT && (MODE == 0 || MODE == 3)) {
        float cinv_col = 0.0f;
        if constexpr (MODE == 3) cinv_col = cinv[(long long)b * N_ + col];
        v8bf o;
#pragma unroll
        for (int v = 0; v < 8; ++v) {
          float val = acc[mt][t][v];
          if constexpr (MODE == 3) {
            const int row = rbase + v;
            val = xres[(long long)row * XRS + col + (long long)b * XBS] - val * cinv_col;
          }
          o[v] = (__bf16)val;
        }
        *(v8bf*)((__bf16*)Cm + (long long)col * CRS + rbase + (long long)b * CBS) = o;
      } else {
#pragma unroll
        for (int v = 0; v < 8; ++v) {
          const int row = rbase + v;
          const long long cidx = CT
              ? ((long long)col * CRS + row + (long long)b * CBS)
              : ((long long)row * CRS + col + (long long)b * CBS);
          const float val = acc[mt][t][v];
          if constexpr (MODE == 0) {
            ((__bf16*)Cm)[cidx] = (__bf16)val;
          } else if constexpr (MODE == 1) {
            ((__bf16*)Cm)[cidx] = (__bf16)(val + bias[row]);
          } else if constexpr (MODE == 2) {
            ((float*)Cm)[cidx] = val;
          } else if constexpr (MODE == 3) {
            const float xr = val * cinv[(long long)b * N_ + col];
            ((__bf16*)Cm)[cidx] =
                (__bf16)(xres[(long long)row * XRS + col + (long long)b * XBS] - xr);
          } else {
            float h = val + bias[row];
            const float inv = gamma[row] * rsqrtf(var[row] + BN_EPS);
            h = h * inv + (beta[row] - mean[row] * inv);
            h = h > 0.0f ? h : 0.0f;
            h += xres[(long long)row * XRS + col + (long long)b * XBS];
            ((float*)Cm)[cidx] = h;
          }
        }
      }
    }
  }
}

// ---------------------------------------------------------------------------
// LDS-tiled transpose + convert-to-bf16.  in: [R][Cc] (f32 or bf16, row major)
// out: [Cc][R] bf16.  32x32 tiles, 256 threads (32x8), coalesced both ways.
// ---------------------------------------------------------------------------
template<bool IN_F32>
__global__ __launch_bounds__(256) void oa_transpose(const void* __restrict__ in,
                                                    __bf16* __restrict__ out,
                                                    int R, int Cc) {
  __shared__ __bf16 tile[32][33];
  const long long base = (long long)blockIdx.z * R * Cc;
  const int c0 = blockIdx.x * 32, r0 = blockIdx.y * 32;
  const int tx = threadIdx.x & 31, ty = threadIdx.x >> 5;
#pragma unroll
  for (int i = 0; i < 4; ++i) {
    const int r = r0 + ty + 8 * i;
    const long long idx = base + (long long)r * Cc + c0 + tx;
    const float v = IN_F32 ? ((const float*)in)[idx]
                           : (float)((const __bf16*)in)[idx];
    tile[ty + 8 * i][tx] = (__bf16)v;
  }
  __syncthreads();
#pragma unroll
  for (int i = 0; i < 4; ++i) {
    const int c = c0 + ty + 8 * i;
    out[base + (long long)c * R + r0 + tx] = tile[tx][ty + 8 * i];
  }
}

// Flat f32 -> bf16 convert (weights)
__global__ __launch_bounds__(256) void oa_cvt(const float* __restrict__ in,
                                              __bf16* __restrict__ out, int n) {
  const int i = blockIdx.x * 256 + threadIdx.x;
  if (i < n) out[i] = (__bf16)in[i];
}

// ---------------------------------------------------------------------------
// Row softmax: attn[b,n,:] = softmax(energy[b,n,:]) as bf16 (row-major)
// ---------------------------------------------------------------------------
__global__ __launch_bounds__(256) void oa_softmax(const float* __restrict__ energy,
                                                  __bf16* __restrict__ attn) {
  __shared__ float red[256];
  const long long row = blockIdx.x;              // b*N + n
  const float* e = energy + row * N_;
  __bf16* a = attn + row * N_;
  const int t = threadIdx.x;

  float lmax = -3.4e38f;
  for (int m = t; m < N_; m += 256) lmax = fmaxf(lmax, e[m]);
  red[t] = lmax; __syncthreads();
  for (int s = 128; s > 0; s >>= 1) { if (t < s) red[t] = fmaxf(red[t], red[t + s]); __syncthreads(); }
  const float mx = red[0]; __syncthreads();

  float lsum = 0.0f;
  for (int m = t; m < N_; m += 256) lsum += __expf(e[m] - mx);
  red[t] = lsum; __syncthreads();
  for (int s = 128; s > 0; s >>= 1) { if (t < s) red[t] += red[t + s]; __syncthreads(); }
  const float inv = 1.0f / red[0];

  for (int m = t; m < N_; m += 256) a[m] = (__bf16)(__expf(e[m] - mx) * inv);
}

// ---------------------------------------------------------------------------
// Column sums over query axis: cinv[b,m] = 1/(1e-9 + sum_n attn[b,n,m])
// ---------------------------------------------------------------------------
__global__ __launch_bounds__(256) void oa_colsum(const __bf16* __restrict__ attn,
                                                 float* __restrict__ cinv) {
  const long long gid = (long long)blockIdx.x * 256 + threadIdx.x;   // over B*N
  const long long b = gid / N_;
  const long long m = gid % N_;
  const __bf16* a = attn + b * (long long)N_ * N_ + m;
  float s = 0.0f;
  for (int n = 0; n < N_; ++n) s += (float)a[(long long)n * N_];
  cinv[gid] = 1.0f / (1e-9f + s);
}

// ---------------------------------------------------------------------------
extern "C" void kernel_launch(void* const* d_in, const int* in_sizes, int n_in,
                              void* d_out, int out_size, void* d_ws, size_t ws_size,
                              hipStream_t stream) {
  (void)in_sizes; (void)n_in; (void)out_size; (void)ws_size;
  const float* q     = (const float*)d_in[0];   // [B, C, N]
  const float* x     = (const float*)d_in[1];   // [B, C2, N]
  const float* Wq    = (const float*)d_in[2];   // [C2, C]
  const float* Wk    = (const float*)d_in[3];   // [C2, C2]
  const float* Wv    = (const float*)d_in[4];   // [C2, C]
  const float* bv    = (const float*)d_in[5];
  const float* Wt    = (const float*)d_in[6];   // [C2, C2]
  const float* bt    = (const float*)d_in[7];
  const float* gamma = (const float*)d_in[8];
  const float* beta  = (const float*)d_in[9];
  const float* rmean = (const float*)d_in[10];
  const float* rvar  = (const float*)d_in[11];
  float* out = (float*)d_out;                   // [B, C2, N]

  char* ws = (char*)d_ws;
  size_t off = 0;
  auto carve = [&](size_t bytes) -> void* {
    void* p = ws + off; off += (bytes + 255) & ~(size_t)255; return p;
  };
  __bf16* qT   = (__bf16*)carve((size_t)B_ * N_ * C_  * 2);  // [B,N,C]  q^T
  __bf16* xT   = (__bf16*)carve((size_t)B_ * N_ * C2_ * 2);  // [B,N,C2] x^T
  __bf16* Wqb  = (__bf16*)carve((size_t)C2_ * C_  * 2);
  __bf16* Wkb  = (__bf16*)carve((size_t)C2_ * C2_ * 2);
  __bf16* Wvb  = (__bf16*)carve((size_t)C2_ * C_  * 2);
  __bf16* Wtb  = (__bf16*)carve((size_t)C2_ * C2_ * 2);
  __bf16* xq   = (__bf16*)carve((size_t)B_ * N_ * C2_ * 2); // [B,N,C2] x_q (n-major)
  __bf16* xkT  = (__bf16*)carve((size_t)B_ * N_ * C2_ * 2); // [B,N,C2] x_k^T
  __bf16* xv   = (__bf16*)carve((size_t)B_ * C2_ * N_ * 2); // [B,C2,N] x_v
  __bf16* ddT  = (__bf16*)carve((size_t)B_ * N_ * C2_ * 2); // [B,N,C2] (x-x_r)^T
  float*  cinv = (float*) carve((size_t)B_ * N_ * 4);
  float*  eng  = (float*) carve((size_t)B_ * N_ * N_ * 4);  // [B,N,N]
  __bf16* attn = (__bf16*)carve((size_t)B_ * N_ * N_ * 2);  // [B,N,N]
  __bf16* attnT= (__bf16*)carve((size_t)B_ * N_ * N_ * 2);  // [B,N,N] attn^T

  const dim3 blk(128);

  // ---- one-time layout conversion (bf16 + K-contiguous layouts) ----
  oa_transpose<true><<<dim3(N_ / 32, C_ / 32, B_), dim3(256), 0, stream>>>(q, qT, C_, N_);
  oa_transpose<true><<<dim3(N_ / 32, C2_ / 32, B_), dim3(256), 0, stream>>>(x, xT, C2_, N_);
  oa_cvt<<<dim3((C2_ * C_ + 255) / 256), dim3(256), 0, stream>>>(Wq, Wqb, C2_ * C_);
  oa_cvt<<<dim3((C2_ * C2_ + 255) / 256), dim3(256), 0, stream>>>(Wk, Wkb, C2_ * C2_);
  oa_cvt<<<dim3((C2_ * C_ + 255) / 256), dim3(256), 0, stream>>>(Wv, Wvb, C2_ * C_);
  oa_cvt<<<dim3((C2_ * C2_ + 255) / 256), dim3(256), 0, stream>>>(Wt, Wtb, C2_ * C2_);

  // 1) x_q[n][o] = sum_c Wq[o,c] q[c,n]; A=Wqb, B=qT, CT store -> xq [n][o]
  oa_gemm<0, true, C_, 0LL, C_, (long long)N_ * C_,
          C2_, (long long)N_ * C2_, 0LL, 0LL, C_>
      <<<dim3(C2_ / 32, N_ / 256, B_), blk, 0, stream>>>(
          Wqb, qT, xq, nullptr, nullptr, nullptr, nullptr, nullptr, nullptr, nullptr);

  // 2) x_k^T[m][o] = sum_c Wk[o,c] x[c,m]; A=Wkb, B=xT, CT store -> xkT [m][o]
  oa_gemm<0, true, C2_, 0LL, C2_, (long long)N_ * C2_,
          C2_, (long long)N_ * C2_, 0LL, 0LL, C2_>
      <<<dim3(C2_ / 32, N_ / 256, B_), blk, 0, stream>>>(
          Wkb, xT, xkT, nullptr, nullptr, nullptr, nullptr, nullptr, nullptr, nullptr);

  // 3) x_v[o][m] = sum_c Wv[o,c] q[c,m] + bv[o]; A=Wvb, B=qT, normal store
  oa_gemm<1, false, C_, 0LL, C_, (long long)N_ * C_,
          N_, (long long)C2_ * N_, 0LL, 0LL, C_>
      <<<dim3(C2_ / 32, N_ / 256, B_), blk, 0, stream>>>(
          Wvb, qT, xv, bv, nullptr, nullptr, nullptr, nullptr, nullptr, nullptr);

  // 4) energy[n][m] = sum_c x_q[n,c] x_k[c,m]; A=xq, B=xkT, f32 store
  oa_gemm<2, false, C2_, (long long)N_ * C2_, C2_, (long long)N_ * C2_,
          N_, (long long)N_ * N_, 0LL, 0LL, C2_>
      <<<dim3(N_ / 32, N_ / 256, B_), blk, 0, stream>>>(
          xq, xkT, eng, nullptr, nullptr, nullptr, nullptr, nullptr, nullptr, nullptr);

  // 5) row softmax -> attn (bf16, row-major)
  oa_softmax<<<dim3(B_ * N_), dim3(256), 0, stream>>>(eng, attn);

  // 6) attn^T for K-contiguous B operand of the x_r GEMM
  oa_transpose<false><<<dim3(N_ / 32, N_ / 32, B_), dim3(256), 0, stream>>>(attn, attnT, N_, N_);

  // 7) column sums over query axis -> 1/(1e-9 + colsum)
  oa_colsum<<<dim3(B_ * N_ / 256), dim3(256), 0, stream>>>(attn, cinv);

  // 8) d^T[m][c] = x[c,m] - (sum_n x_v[c,n] attn[n,m]) * cinv[m]; CT store
  oa_gemm<3, true, N_, (long long)C2_ * N_, N_, (long long)N_ * N_,
          C2_, (long long)N_ * C2_, N_, (long long)C2_ * N_, N_>
      <<<dim3(C2_ / 32, N_ / 256, B_), blk, 0, stream>>>(
          xv, attnT, ddT, nullptr, cinv, x, nullptr, nullptr, nullptr, nullptr);

  // 9) out[o][m] = relu(BN(sum_c Wt[o,c] d[c,m] + bt[o])) + x[o,m]
  oa_gemm<4, false, C2_, 0LL, C2_, (long long)N_ * C2_,
          N_, (long long)C2_ * N_, N_, (long long)C2_ * N_, C2_>
      <<<dim3(C2_ / 32, N_ / 256, B_), blk, 0, stream>>>(
          Wtb, ddT, out, bt, nullptr, x, gamma, beta, rmean, rvar);
}
